// EdgeWeight_62027917688907
// MI455X (gfx1250) — matrix operand order repaired and checked by reading, not compiled, not run on hardware
//
#include <hip/hip_runtime.h>
#include <hip/hip_bf16.h>

typedef __attribute__((ext_vector_type(16))) __bf16 v16bf;
typedef __attribute__((ext_vector_type(8)))  float  v8f;
typedef __attribute__((ext_vector_type(4)))  float  v4f;

#define DEGREE          32
#define EDGE_DIM        128
#define HID             64
#define TOPK            8
#define WAVES_PER_BLOCK 8
#define THREADS         (WAVES_PER_BLOCK * 32)
#define B_ELEMS         (16 * 32 * 16)   // 16 fragments x 32 lanes x 16 bf16 = all of W1

__global__ __launch_bounds__(THREADS)
void edge_weight_topk_kernel(const float* __restrict__ e_feats,
                             const float* __restrict__ W1,
                             const float* __restrict__ b1,
                             const float* __restrict__ W2,
                             float* __restrict__ out,
                             int n_nodes)
{
    // W1 pre-swizzled into WMMA B-fragment layout (bf16), 16 KB.
    __shared__ __attribute__((aligned(32))) __bf16 sB[B_ELEMS];
    __shared__ float sScores[WAVES_PER_BLOCK][DEGREE];

    const int tid = threadIdx.x;

    // ---- one-time block preload: W1 (fp32, [128][64]) -> bf16 B fragments ----
    // Fragment (t, nt): B tile = W1[32t .. 32t+31][16nt .. 16nt+15]
    // B 32x16 layout used here (contiguous k, not ISA table order): lanes 0-15 hold K=0..15 (col = lane),
    //                            lanes 16-31 hold K=16..31.
    for (int i = tid; i < B_ELEMS; i += THREADS) {
        int e    = i & 15;          // element within lane's 16 bf16
        int lane = (i >> 4) & 31;
        int fi   = i >> 9;          // fragment index 0..15
        int t    = fi >> 2;         // K-step
        int nt   = fi & 3;          // N-tile
        int k    = 32 * t + 16 * (lane >> 4) + e;
        int col  = nt * 16 + (lane & 15);
        sB[i] = (__bf16)W1[k * HID + col];
    }
    __syncthreads();

    const int lane    = tid & 31;
    const int w       = tid >> 5;
    const int node    = blockIdx.x * WAVES_PER_BLOCK + w;
    if (node >= n_nodes) return;

    const int lmod    = lane & 15;
    const int halfsel = lane >> 4;    // 0: lanes 0-15, 1: lanes 16-31

    // Per-lane slice of bias / W2: column n = nt*16 + (lane&15)  (matches C layout N=lane)
    float b1v[4], w2v[4];
#pragma unroll
    for (int nt = 0; nt < 4; ++nt) {
        int n = nt * 16 + lmod;
        b1v[nt] = b1[n];
        w2v[nt] = W2[n];
    }

    // ---- two 16-edge M-tiles per node: h = relu(X*W1 + b1); score = h . W2 ----
#pragma unroll
    for (int mt = 0; mt < 2; ++mt) {
        const int row = node * DEGREE + mt * 16 + lmod;   // A layout: M = lane&15
        const v4f* rp = (const v4f*)(e_feats + (size_t)row * EDGE_DIM);

        v8f acc[4];
#pragma unroll
        for (int nt = 0; nt < 4; ++nt) { v8f z = {}; acc[nt] = z; }

#pragma unroll
        for (int t = 0; t < 4; ++t) {
            // A 16x32 bf16 layout: this lane half needs K run [k0,k0+15] (contiguous, matches sB)
            const int k0q = (32 * t + 16 * halfsel) >> 2;    // in float4 units
            v4f f0 = __builtin_nontemporal_load(rp + k0q + 0);
            v4f f1 = __builtin_nontemporal_load(rp + k0q + 1);
            v4f f2 = __builtin_nontemporal_load(rp + k0q + 2);
            v4f f3 = __builtin_nontemporal_load(rp + k0q + 3);
            v16bf a;
            a[0]  = (__bf16)f0[0]; a[1]  = (__bf16)f0[1]; a[2]  = (__bf16)f0[2]; a[3]  = (__bf16)f0[3];
            a[4]  = (__bf16)f1[0]; a[5]  = (__bf16)f1[1]; a[6]  = (__bf16)f1[2]; a[7]  = (__bf16)f1[3];
            a[8]  = (__bf16)f2[0]; a[9]  = (__bf16)f2[1]; a[10] = (__bf16)f2[2]; a[11] = (__bf16)f2[3];
            a[12] = (__bf16)f3[0]; a[13] = (__bf16)f3[1]; a[14] = (__bf16)f3[2]; a[15] = (__bf16)f3[3];

#pragma unroll
            for (int nt = 0; nt < 4; ++nt) {
                const v16bf bfrag =
                    *(const v16bf*)&sB[(((t * 4 + nt) * 32) + lane) * 16];
                acc[nt] = __builtin_amdgcn_wmma_f32_16x16x32_bf16(
                    false, a, false, bfrag, (short)0, acc[nt], false, false);
            }
        }

        // Epilogue: C layout -> VGPR r holds row (r + 8*halfsel), N = lane&15.
        // score[m] = sum_n relu(h[m,n] + b1[n]) * W2[n]; reduce N across the 16-lane half.
#pragma unroll
        for (int r = 0; r < 8; ++r) {
            float p = 0.0f;
#pragma unroll
            for (int nt = 0; nt < 4; ++nt) {
                float h = acc[nt][r] + b1v[nt];
                h = fmaxf(h, 0.0f);
                p = fmaf(h, w2v[nt], p);
            }
            p += __shfl_xor(p, 1, 32);
            p += __shfl_xor(p, 2, 32);
            p += __shfl_xor(p, 4, 32);
            p += __shfl_xor(p, 8, 32);
            if (lmod == r)
                sScores[w][mt * 16 + r + 8 * halfsel] = p;
        }
    }

    // ---- per-node softmax over 32 edges + stable top-8 mask (keep self-loop 0) ----
    // (b2 omitted: softmax is shift-invariant.)
    float s = sScores[w][lane];
    float m = s;
#pragma unroll
    for (int msk = 16; msk >= 1; msk >>= 1) m = fmaxf(m, __shfl_xor(m, msk, 32));
    float p = __expf(s - m);
    float sum = p;
#pragma unroll
    for (int msk = 16; msk >= 1; msk >>= 1) sum += __shfl_xor(sum, msk, 32);
    float wgt = p / sum;

    int cnt = 0;
#pragma unroll
    for (int j = 0; j < 32; ++j) {
        float wj = __shfl(wgt, j, 32);
        cnt += ((wj > wgt) || ((wj == wgt) && (j < lane))) ? 1 : 0;
    }
    float res = ((cnt < TOPK) || (lane == 0)) ? wgt : 0.0f;
    out[(size_t)node * DEGREE + lane] = res;
}

extern "C" void kernel_launch(void* const* d_in, const int* in_sizes, int n_in,
                              void* d_out, int out_size, void* d_ws, size_t ws_size,
                              hipStream_t stream) {
    const float* e_feats = (const float*)d_in[0];
    const float* W1      = (const float*)d_in[1];
    const float* b1      = (const float*)d_in[2];
    const float* W2      = (const float*)d_in[3];
    // d_in[4] = b2: unused (softmax shift-invariance)
    const int n_nodes = in_sizes[0] / (DEGREE * EDGE_DIM);
    const int blocks  = (n_nodes + WAVES_PER_BLOCK - 1) / WAVES_PER_BLOCK;
    hipLaunchKernelGGL(edge_weight_topk_kernel, dim3(blocks), dim3(THREADS), 0, stream,
                       e_feats, W1, b1, W2, (float*)d_out, n_nodes);
}
